// MOEAdapterBlock_83880711291356
// MI455X (gfx1250) — matrix operand order repaired and checked by reading, not compiled, not run on hardware
//
#include <hip/hip_runtime.h>

// ---------------------------------------------------------------------------
// Types for CDNA5 WMMA (gfx1250, wave32)
// ---------------------------------------------------------------------------
typedef __attribute__((ext_vector_type(16))) __bf16 bf16x16;
typedef __attribute__((ext_vector_type(8)))  float  f32x8;

#define DIMC     768
#define HEADSC   12
#define DHEAD    64
#define MLPH     3072
#define NTOK     8192          // B*N = 8*1024
#define SEQN     1024
#define BATCH    8
#define EXP_DH   384

// ---------------------------------------------------------------------------
// CDNA5 async global->LDS (ASYNCcnt) with sync fallback
// ---------------------------------------------------------------------------
#if defined(__has_builtin)
#  if __has_builtin(__builtin_amdgcn_global_load_async_to_lds_b128)
#    define HAVE_ASYNC_LDS 1
#  endif
#  if __has_builtin(__builtin_amdgcn_s_wait_asynccnt)
#    define HAVE_ASYNC_WAIT 1
#  endif
#endif
#ifndef HAVE_ASYNC_LDS
#  define HAVE_ASYNC_LDS 0
#endif
#ifndef HAVE_ASYNC_WAIT
#  define HAVE_ASYNC_WAIT 0
#endif

// GCC-style int4 vector: matches the builtin's parameter pointee type exactly.
typedef int v4i_vs __attribute__((vector_size(4 * sizeof(int))));

template <int OFF>
__device__ __forceinline__ void async_copy16(const __bf16* g, __bf16* l) {
#if HAVE_ASYNC_LDS
    __builtin_amdgcn_global_load_async_to_lds_b128(
        (v4i_vs __attribute__((address_space(1)))*)(g),
        (v4i_vs __attribute__((address_space(3)))*)(l), OFF, 0);
#else
    uint4 v = *(const uint4*)((const char*)g + OFF);
    *(uint4*)((char*)l + OFF) = v;
#endif
}

__device__ __forceinline__ void wait_async_zero() {
#if HAVE_ASYNC_WAIT
    __builtin_amdgcn_s_wait_asynccnt(0);
#elif HAVE_ASYNC_LDS
    asm volatile("s_wait_asynccnt 0x0" ::: "memory");
#endif
}

// ---------------------------------------------------------------------------
// Helpers
// ---------------------------------------------------------------------------
__device__ __forceinline__ __bf16 f2bf(float f) {
    union { float f; unsigned u; } a; a.f = f;
    unsigned r = a.u + 0x7FFFu + ((a.u >> 16) & 1u);   // round-to-nearest-even
    unsigned short h = (unsigned short)(r >> 16);
    return __builtin_bit_cast(__bf16, h);
}
__device__ __forceinline__ float bf2f(__bf16 b) {
    unsigned short h = __builtin_bit_cast(unsigned short, b);
    union { unsigned u; float f; } a; a.u = ((unsigned)h) << 16;
    return a.f;
}
__device__ __forceinline__ float gelu_exact(float v) {
    return 0.5f * v * (1.0f + erff(v * 0.70710678118654752f));
}
__device__ __forceinline__ f32x8 vzero8() {
    f32x8 v;
#pragma unroll
    for (int i = 0; i < 8; ++i) v[i] = 0.0f;
    return v;
}

// A-fragment (16x32 bf16, M x K): lane m=l&15, h=l>>4.
// elements 0..15 -> K = h*16 + e
__device__ __forceinline__ bf16x16 load_A_frag(const __bf16* a, int lda,
                                               int row0, int k0, int lane) {
    const int m = lane & 15, h = lane >> 4;
    const __bf16* p = a + (size_t)(row0 + m) * lda + k0;
    bf16x16 r;
#pragma unroll
    for (int e = 0; e < 8; ++e) r[e] = p[h * 16 + e];
#pragma unroll
    for (int e = 0; e < 8; ++e) r[8 + e] = p[h * 16 + 8 + e];
    return r;
}

// B-fragment (32x16 bf16, K x N) from K-contiguous (transposed) buffer:
// bt[(col)*ldk + k]. lanes 0-15: K=0..15, lanes 16-31: K=16..31.
__device__ __forceinline__ bf16x16 load_B_fragT(const __bf16* bt, int ldk,
                                                int col0, int k0, int lane) {
    const int n = lane & 15, h = lane >> 4;
    const __bf16* p = bt + (size_t)(col0 + n) * ldk + k0 + h * 16;
    bf16x16 r;
#pragma unroll
    for (int e = 0; e < 16; ++e) r[e] = p[e];
    return r;
}

// ---------------------------------------------------------------------------
// f32 -> bf16 conversion
// ---------------------------------------------------------------------------
__global__ __launch_bounds__(256)
void k_f32_to_bf16(const float* __restrict__ in, __bf16* __restrict__ out, size_t n) {
    size_t i = (size_t)blockIdx.x * 256 + threadIdx.x;
    if (i < n) out[i] = f2bf(in[i]);
}

// ---------------------------------------------------------------------------
// LayerNorm (one block per token), bf16 output
// ---------------------------------------------------------------------------
__global__ __launch_bounds__(256)
void k_layernorm_bf16(const float* __restrict__ x, const float* __restrict__ w,
                      const float* __restrict__ b, __bf16* __restrict__ out) {
    const int token = blockIdx.x;
    const float* row = x + (size_t)token * DIMC;
    __shared__ float red[256];
    const int tid = threadIdx.x;

    float s = 0.f;
    for (int i = tid; i < DIMC; i += 256) s += row[i];
    red[tid] = s; __syncthreads();
    for (int off = 128; off > 0; off >>= 1) {
        if (tid < off) red[tid] += red[tid + off];
        __syncthreads();
    }
    const float mu = red[0] / (float)DIMC;
    __syncthreads();

    float vs = 0.f;
    for (int i = tid; i < DIMC; i += 256) { float d = row[i] - mu; vs += d * d; }
    red[tid] = vs; __syncthreads();
    for (int off = 128; off > 0; off >>= 1) {
        if (tid < off) red[tid] += red[tid + off];
        __syncthreads();
    }
    const float rstd = rsqrtf(red[0] / (float)DIMC + 1e-5f);

    for (int i = tid; i < DIMC; i += 256)
        out[(size_t)token * DIMC + i] = f2bf((row[i] - mu) * rstd * w[i] + b[i]);
}

// ---------------------------------------------------------------------------
// Generic bf16 WMMA GEMM: out = epi(A[MxK] @ B[KxN])
// FLAGS: 1=bias 2=gelu 4=residual-add 8=scale-from-memory 16=bf16-store
// epilogue: v = acc (+bias); (gelu); v *= scale; (v += res); store
// 256 thr = 8 waves (2Mx4N), tile 128x128, K-step 32, double-buffered LDS,
// A tiles staged via async global->LDS, B tiles register-staged + transposed.
// ---------------------------------------------------------------------------
template <int FLAGS>
__global__ __launch_bounds__(256)
void k_gemm_bf16_wmma(const __bf16* __restrict__ A, const __bf16* __restrict__ Bw,
                      const float* __restrict__ bias, const float* __restrict__ res,
                      const float* __restrict__ scale_p, void* __restrict__ outv,
                      int M, int N, int K) {
    constexpr bool HAS_BIAS  = (FLAGS & 1)  != 0;
    constexpr bool DO_GELU   = (FLAGS & 2)  != 0;
    constexpr bool HAS_RES   = (FLAGS & 4)  != 0;
    constexpr bool HAS_SCALE = (FLAGS & 8)  != 0;
    constexpr bool OUT_BF16  = (FLAGS & 16) != 0;

    __shared__ __bf16 sA[2][128 * 32];    // [row][k]
    __shared__ __bf16 sBT[2][128 * 32];   // [col][k] (transposed)

    const int tid  = threadIdx.x;
    const int lane = tid & 31, wave = tid >> 5;
    const int wm   = wave >> 2, wn = wave & 3;       // 2 x 4 waves
    const int m0   = blockIdx.y * 128, n0 = blockIdx.x * 128;

    f32x8 acc[4][2];
#pragma unroll
    for (int i = 0; i < 4; ++i)
#pragma unroll
        for (int j = 0; j < 2; ++j) acc[i][j] = vzero8();

    const int arow = tid >> 1, aseg = tid & 1;   // 128 rows x 2 halves of 16
    const int bkr  = tid >> 3, bseg = tid & 7;   // 32 k-rows x 8 groups of 16 cols

    // stage A tile (async): 16 bf16 (32B) per thread
    auto stage_A = [&](int buf, int kk) {
        const __bf16* ga = A + (size_t)(m0 + arow) * K + kk + aseg * 16;
        __bf16* la = &sA[buf][arow * 32 + aseg * 16];
        async_copy16<0>(ga, la);
        async_copy16<16>(ga, la);
    };
    auto load_B_regs = [&](int kk, uint4& b0, uint4& b1) {
        const __bf16* gb = Bw + (size_t)(kk + bkr) * N + n0 + bseg * 16;
        b0 = ((const uint4*)gb)[0];
        b1 = ((const uint4*)gb)[1];
    };
    auto store_BT = [&](int buf, uint4 b0, uint4 b1) {
        const __bf16* p0 = (const __bf16*)&b0;
        const __bf16* p1 = (const __bf16*)&b1;
#pragma unroll
        for (int j = 0; j < 8; ++j) sBT[buf][(bseg * 16 + j) * 32 + bkr] = p0[j];
#pragma unroll
        for (int j = 0; j < 8; ++j) sBT[buf][(bseg * 16 + 8 + j) * 32 + bkr] = p1[j];
    };

    // ---- prologue: stage first tile
    {
        uint4 b0, b1;
        stage_A(0, 0);
        load_B_regs(0, b0, b1);
        store_BT(0, b0, b1);
        wait_async_zero();
    }
    __syncthreads();

    int buf = 0;
    for (int kk = 0; kk < K; kk += 32) {
        const bool more = (kk + 32) < K;
        uint4 b0, b1;
        if (more) {
            stage_A(buf ^ 1, kk + 32);        // async into next buffer
            load_B_regs(kk + 32, b0, b1);     // global loads issued early
        }

        // ---- compute on current buffer: 8 WMMA
        bf16x16 af[4], bfr[2];
#pragma unroll
        for (int i = 0; i < 4; ++i)
            af[i] = load_A_frag(sA[buf], 32, wm * 64 + i * 16, 0, lane);
#pragma unroll
        for (int j = 0; j < 2; ++j)
            bfr[j] = load_B_fragT(sBT[buf], 32, wn * 32 + j * 16, 0, lane);
#pragma unroll
        for (int i = 0; i < 4; ++i)
#pragma unroll
            for (int j = 0; j < 2; ++j)
                acc[i][j] = __builtin_amdgcn_wmma_f32_16x16x32_bf16(
                    false, af[i], false, bfr[j], (short)0, acc[i][j], false, false);

        if (more) {
            store_BT(buf ^ 1, b0, b1);
            wait_async_zero();
        }
        __syncthreads();
        buf ^= 1;
    }

    // ---- epilogue (C layout: row = r + 8*half, col = lane&15)
    const int cn = lane & 15, ch = lane >> 4;
    const float scale = HAS_SCALE ? *scale_p : 1.0f;
#pragma unroll
    for (int i = 0; i < 4; ++i) {
#pragma unroll
        for (int j = 0; j < 2; ++j) {
            const int col = n0 + wn * 32 + j * 16 + cn;
            const float bcol = HAS_BIAS ? bias[col] : 0.0f;
#pragma unroll
            for (int r = 0; r < 8; ++r) {
                const int row = m0 + wm * 64 + i * 16 + ch * 8 + r;
                float v = acc[i][j][r] + bcol;
                if (DO_GELU) v = gelu_exact(v);
                v *= scale;
                if (HAS_RES) v += res[(size_t)row * N + col];
                if (OUT_BF16) ((__bf16*)outv)[(size_t)row * N + col] = f2bf(v);
                else          ((float*)outv)[(size_t)row * N + col] = v;
            }
        }
    }
}

// ---------------------------------------------------------------------------
// Flash attention: grid (16 q-blocks, 12 heads, 8 batch), 128 thr = 4 waves.
// 64-key blocks: 16 WMMA per iteration; K staged via async global->LDS.
// Q bf16 (B,N,768) head-strided; KV bf16 (B,N,1536): K at h*64, V at 768+h*64.
// ---------------------------------------------------------------------------
__global__ __launch_bounds__(128)
void k_attention_wmma(const __bf16* __restrict__ Q, const __bf16* __restrict__ KV,
                      __bf16* __restrict__ Out) {
    const int b = blockIdx.z, hh = blockIdx.y;
    const int q0 = blockIdx.x * 64;
    const int tid = threadIdx.x, lane = tid & 31, wave = tid >> 5;
    const float sm_scale = 0.125f;   // 1/sqrt(64)

    __shared__ __bf16 sK[64 * 64];        // [key][dh]
    __shared__ __bf16 sVT[64 * 64];       // [dh][key]
    __shared__ __bf16 sP[4][16 * 64];     // per-wave P [row][key]

    const __bf16* qbase  = Q  + (size_t)b * SEQN * DIMC + hh * DHEAD;
    const __bf16* kvbase = KV + (size_t)b * SEQN * (2 * DIMC);

    bf16x16 aq0 = load_A_frag(qbase, DIMC, q0 + wave * 16, 0,  lane);
    bf16x16 aq1 = load_A_frag(qbase, DIMC, q0 + wave * 16, 32, lane);

    float m_i[8], l_i[8];
    f32x8 o[4];
#pragma unroll
    for (int r = 0; r < 8; ++r) { m_i[r] = -1e30f; l_i[r] = 0.0f; }
#pragma unroll
    for (int j = 0; j < 4; ++j) o[j] = vzero8();

    const int lkey = tid >> 1, lseg = tid & 1;   // 64 keys x 2 segments of 32 dh

    for (int kb = 0; kb < SEQN; kb += 64) {
        // ---- K rows: async copy 64B per thread straight into LDS
        const __bf16* kg = kvbase + (size_t)(kb + lkey) * (2 * DIMC) + hh * DHEAD + lseg * 32;
        __bf16* kl = sK + lkey * 64 + lseg * 32;
        async_copy16<0>(kg, kl);
        async_copy16<16>(kg, kl);
        async_copy16<32>(kg, kl);
        async_copy16<48>(kg, kl);
        // ---- V rows: register-staged + transposed store
        const __bf16* vg = kg + DIMC;
        uint4 vr[4];
        vr[0] = ((const uint4*)vg)[0];
        vr[1] = ((const uint4*)vg)[1];
        vr[2] = ((const uint4*)vg)[2];
        vr[3] = ((const uint4*)vg)[3];
        const __bf16* vp = (const __bf16*)vr;
#pragma unroll
        for (int e = 0; e < 32; ++e) sVT[(lseg * 32 + e) * 64 + lkey] = vp[e];
        wait_async_zero();
        __syncthreads();

        // ---- S = scale * Q @ K^T  (four 16x16 tiles, 8 WMMA)
        f32x8 s[4];
#pragma unroll
        for (int j = 0; j < 4; ++j) {
            s[j] = vzero8();
            bf16x16 bk0 = load_B_fragT(sK, 64, j * 16, 0,  lane);
            bf16x16 bk1 = load_B_fragT(sK, 64, j * 16, 32, lane);
            s[j] = __builtin_amdgcn_wmma_f32_16x16x32_bf16(false, aq0, false, bk0, (short)0, s[j], false, false);
            s[j] = __builtin_amdgcn_wmma_f32_16x16x32_bf16(false, aq1, false, bk1, (short)0, s[j], false, false);
#pragma unroll
            for (int r = 0; r < 8; ++r) s[j][r] *= sm_scale;
        }

        // ---- online softmax (row r lives across the lane's 16-lane half)
#pragma unroll
        for (int r = 0; r < 8; ++r) {
            float mr = fmaxf(fmaxf(s[0][r], s[1][r]), fmaxf(s[2][r], s[3][r]));
#pragma unroll
            for (int mask = 1; mask < 16; mask <<= 1)
                mr = fmaxf(mr, __shfl_xor(mr, mask, 32));
            const float mnew = fmaxf(m_i[r], mr);
            const float corr = __expf(m_i[r] - mnew);
            l_i[r] *= corr;
#pragma unroll
            for (int j2 = 0; j2 < 4; ++j2) o[j2][r] *= corr;
            float rs = 0.0f;
#pragma unroll
            for (int j = 0; j < 4; ++j) {
                const float p = __expf(s[j][r] - mnew);
                s[j][r] = p;
                rs += p;
            }
#pragma unroll
            for (int mask = 1; mask < 16; mask <<= 1)
                rs += __shfl_xor(rs, mask, 32);
            l_i[r] += rs;
            m_i[r] = mnew;
        }

        // ---- spill P (C layout) to LDS row-major, reload in A layout
        {
            const int cn = lane & 15, ch = lane >> 4;
#pragma unroll
            for (int j = 0; j < 4; ++j)
#pragma unroll
                for (int r = 0; r < 8; ++r)
                    sP[wave][(ch * 8 + r) * 64 + j * 16 + cn] = f2bf(s[j][r]);
        }
        __syncthreads();

        bf16x16 ap0 = load_A_frag(&sP[wave][0], 64, 0, 0,  lane);
        bf16x16 ap1 = load_A_frag(&sP[wave][0], 64, 0, 32, lane);
#pragma unroll
        for (int j2 = 0; j2 < 4; ++j2) {
            bf16x16 bv0 = load_B_fragT(sVT, 64, j2 * 16, 0,  lane);
            bf16x16 bv1 = load_B_fragT(sVT, 64, j2 * 16, 32, lane);
            o[j2] = __builtin_amdgcn_wmma_f32_16x16x32_bf16(false, ap0, false, bv0, (short)0, o[j2], false, false);
            o[j2] = __builtin_amdgcn_wmma_f32_16x16x32_bf16(false, ap1, false, bv1, (short)0, o[j2], false, false);
        }
        __syncthreads();
    }

    // ---- normalize and store bf16 (B,N,768) with col = h*64 + d
    const int cn = lane & 15, ch = lane >> 4;
#pragma unroll
    for (int j2 = 0; j2 < 4; ++j2)
#pragma unroll
        for (int r = 0; r < 8; ++r) {
            const int row = q0 + wave * 16 + ch * 8 + r;
            const int col = hh * DHEAD + j2 * 16 + cn;
            Out[(size_t)(b * SEQN + row) * DIMC + col] = f2bf(o[j2][r] / l_i[r]);
        }
}

// ---------------------------------------------------------------------------
// Depthwise 3x3 conv (SAME, 32x32 spatial) + bias + exact GELU, bf16 in/out.
// ---------------------------------------------------------------------------
__global__ __launch_bounds__(256)
void k_dwconv_gelu(const __bf16* __restrict__ in, const float* __restrict__ w9,
                   const float* __restrict__ bias, __bf16* __restrict__ out) {
    const size_t total = (size_t)NTOK * MLPH;
    size_t idx = (size_t)blockIdx.x * 256 + threadIdx.x;
    if (idx >= total) return;
    const int c = (int)(idx % MLPH);
    const size_t t = idx / MLPH;
    const int n = (int)(t % SEQN);
    const int b = (int)(t / SEQN);
    const int y = n >> 5, x = n & 31;

    float acc = bias[c];
#pragma unroll
    for (int dy = -1; dy <= 1; ++dy) {
        const int yy = y + dy;
        if (yy < 0 || yy > 31) continue;
#pragma unroll
        for (int dx = -1; dx <= 1; ++dx) {
            const int xx = x + dx;
            if (xx < 0 || xx > 31) continue;
            const size_t src = ((size_t)b * SEQN + yy * 32 + xx) * MLPH + c;
            acc += w9[c * 9 + (dy + 1) * 3 + (dx + 1)] * bf2f(in[src]);
        }
    }
    out[idx] = f2bf(gelu_exact(acc));
}

// ---------------------------------------------------------------------------
// Gating: per-block partial sums of noisy logits over 256 tokens.
// ---------------------------------------------------------------------------
__global__ __launch_bounds__(256)
void k_gate_partial(const float* __restrict__ x, const float* __restrict__ gate_w,
                    const int* __restrict__ label, const float* __restrict__ eps,
                    float* __restrict__ part) {
    const int blk = blockIdx.x;
    const int token = blk * 256 + threadIdx.x;
    const float* gw = gate_w + (size_t)label[0] * DIMC * 16;
    const float* row = x + (size_t)token * DIMC;

    float tw[16];
#pragma unroll
    for (int j = 0; j < 16; ++j) tw[j] = 0.0f;
    for (int d = 0; d < DIMC; ++d) {
        const float xv = row[d];
#pragma unroll
        for (int j = 0; j < 16; ++j) tw[j] += xv * gw[d * 16 + j];
    }
    float lg[8];
#pragma unroll
    for (int e = 0; e < 8; ++e) {
        const float raw = tw[8 + e];
        const float sp = (raw > 20.0f) ? raw : log1pf(__expf(raw));
        lg[e] = tw[e] + eps[(size_t)token * 8 + e] * (sp + 0.01f);
    }

    __shared__ float red[256];
    const int tid = threadIdx.x;
    for (int e = 0; e < 8; ++e) {
        red[tid] = lg[e]; __syncthreads();
        for (int off = 128; off > 0; off >>= 1) {
            if (tid < off) red[tid] += red[tid + off];
            __syncthreads();
        }
        if (tid == 0) part[blk * 8 + e] = red[0];
        __syncthreads();
    }
}

// exp_wise_sum, MI_loss, top-2, global min/max, batch-0 softmax -> gates[2]
__global__ void k_gate_final(const float* __restrict__ part, float* __restrict__ mi_out,
                             float* __restrict__ gates) {
    if (threadIdx.x != 0 || blockIdx.x != 0) return;
    float es[BATCH][8];
    for (int b = 0; b < BATCH; ++b)
        for (int e = 0; e < 8; ++e) {
            float s = 0.f;
            for (int p = 0; p < 4; ++p) s += part[(b * 4 + p) * 8 + e];
            es[b][e] = s;
            mi_out[b * 8 + e] = s / 1024.0f;   // H*W
        }
    float t0[BATCH], t1[BATCH];
    for (int b = 0; b < BATCH; ++b) {
        float v0 = -1e30f, v1 = -1e30f;
        for (int e = 0; e < 8; ++e) {
            const float v = es[b][e];
            if (v > v0) { v1 = v0; v0 = v; }
            else if (v > v1) { v1 = v; }
        }
        t0[b] = v0; t1[b] = v1;
    }
    float m1 = 1e30f, m2 = -1e30f;
    for (int b = 0; b < BATCH; ++b) {
        m1 = fminf(m1, fminf(t0[b], t1[b]));
        m2 = fmaxf(m2, fmaxf(t0[b], t1[b]));
    }
    const float denom = m2 - m1;
    const float a0 = (t0[0] - m1) / denom;
    const float a1 = (t1[0] - m1) / denom;
    const float e0 = __expf(a0), e1 = __expf(a1);
    gates[0] = e0 / (e0 + e1);
    gates[1] = e1 / (e0 + e1);
}

// ---------------------------------------------------------------------------
// Host-side orchestration
// ---------------------------------------------------------------------------
extern "C" void kernel_launch(void* const* d_in, const int* in_sizes, int n_in,
                              void* d_out, int out_size, void* d_ws, size_t ws_size,
                              hipStream_t stream) {
    const float* x      = (const float*)d_in[0];
    const float* n1w    = (const float*)d_in[1];
    const float* n1b    = (const float*)d_in[2];
    const float* n2w    = (const float*)d_in[3];
    const float* n2b    = (const float*)d_in[4];
    const float* q_w    = (const float*)d_in[5];
    const float* kv_w   = (const float*)d_in[6];
    const float* proj_w = (const float*)d_in[7];
    const float* proj_b = (const float*)d_in[8];
    const float* fc1_w  = (const float*)d_in[9];
    const float* fc1_b  = (const float*)d_in[10];
    const float* dw_w   = (const float*)d_in[11];
    const float* dw_b   = (const float*)d_in[12];
    const float* fc2_w  = (const float*)d_in[13];
    const float* fc2_b  = (const float*)d_in[14];
    const float* gate_w = (const float*)d_in[15];
    const float* eps    = (const float*)d_in[16];
    const float* e1w    = (const float*)d_in[17];
    const float* e1b    = (const float*)d_in[18];
    const float* e2w    = (const float*)d_in[19];
    const float* e2b    = (const float*)d_in[20];
    const int*   label  = (const int*)d_in[24];

    // ---- workspace layout
    char* ws = (char*)d_ws;
    size_t off = 0;
    auto take = [&](size_t bytes) -> char* {
        char* p = ws + off;
        off += (bytes + 255) & ~(size_t)255;
        return p;
    };
    __bf16* WQ    = (__bf16*)take((size_t)768 * 768 * 2);
    __bf16* WKV   = (__bf16*)take((size_t)768 * 1536 * 2);
    __bf16* WPROJ = (__bf16*)take((size_t)768 * 768 * 2);
    __bf16* WFC1  = (__bf16*)take((size_t)768 * 3072 * 2);
    __bf16* WFC2  = (__bf16*)take((size_t)3072 * 768 * 2);
    __bf16* WE1   = (__bf16*)take((size_t)2 * 768 * 384 * 2);
    __bf16* WE2   = (__bf16*)take((size_t)2 * 384 * 768 * 2);
    __bf16* HBF   = (__bf16*)take((size_t)NTOK * 768 * 2);
    __bf16* QB    = (__bf16*)take((size_t)NTOK * 768 * 2);
    __bf16* KVB   = (__bf16*)take((size_t)NTOK * 1536 * 2);
    __bf16* AB    = (__bf16*)take((size_t)NTOK * 768 * 2);
    float*  X2    = (float*)take((size_t)NTOK * 768 * 4);
    __bf16* FC1B  = (__bf16*)take((size_t)NTOK * 3072 * 2);
    __bf16* CONVB = (__bf16*)take((size_t)NTOK * 3072 * 2);
    float*  X3    = (float*)take((size_t)NTOK * 768 * 4);
    __bf16* X3BF  = (__bf16*)take((size_t)NTOK * 768 * 2);
    __bf16* EXPH  = (__bf16*)take((size_t)NTOK * 384 * 2);
    float*  GPART = (float*)take(32 * 8 * 4);
    float*  GATES = (float*)take(2 * 4);

    auto cvt = [&](const float* src, __bf16* dst, size_t n) {
        k_f32_to_bf16<<<(unsigned)((n + 255) / 256), 256, 0, stream>>>(src, dst, n);
    };
    // ---- convert weights once per launch
    cvt(q_w,    WQ,    (size_t)768 * 768);
    cvt(kv_w,   WKV,   (size_t)768 * 1536);
    cvt(proj_w, WPROJ, (size_t)768 * 768);
    cvt(fc1_w,  WFC1,  (size_t)768 * 3072);
    cvt(fc2_w,  WFC2,  (size_t)3072 * 768);
    cvt(e1w,    WE1,   (size_t)2 * 768 * 384);   // experts 0,1
    cvt(e2w,    WE2,   (size_t)2 * 384 * 768);

    // ---- attention branch
    k_layernorm_bf16<<<NTOK, 256, 0, stream>>>(x, n1w, n1b, HBF);
    k_gemm_bf16_wmma<16><<<dim3(768 / 128, NTOK / 128), 256, 0, stream>>>(
        HBF, WQ, nullptr, nullptr, nullptr, QB, NTOK, 768, 768);
    k_gemm_bf16_wmma<16><<<dim3(1536 / 128, NTOK / 128), 256, 0, stream>>>(
        HBF, WKV, nullptr, nullptr, nullptr, KVB, NTOK, 1536, 768);
    k_attention_wmma<<<dim3(SEQN / 64, HEADSC, BATCH), 128, 0, stream>>>(QB, KVB, AB);
    k_gemm_bf16_wmma<5><<<dim3(768 / 128, NTOK / 128), 256, 0, stream>>>(
        AB, WPROJ, proj_b, x, nullptr, X2, NTOK, 768, 768);   // X2 = x + a@W + b

    // ---- conv-MLP branch
    k_layernorm_bf16<<<NTOK, 256, 0, stream>>>(X2, n2w, n2b, HBF);
    k_gemm_bf16_wmma<17><<<dim3(3072 / 128, NTOK / 128), 256, 0, stream>>>(
        HBF, WFC1, fc1_b, nullptr, nullptr, FC1B, NTOK, 3072, 768);
    {
        const size_t total = (size_t)NTOK * MLPH;
        k_dwconv_gelu<<<(unsigned)((total + 255) / 256), 256, 0, stream>>>(
            FC1B, dw_w, dw_b, CONVB);
    }
    k_gemm_bf16_wmma<5><<<dim3(768 / 128, NTOK / 128), 256, 0, stream>>>(
        CONVB, WFC2, fc2_b, X2, nullptr, X3, NTOK, 768, 3072); // X3 = X2 + h1@W + b

    // ---- gating (MI_loss written to tail of d_out)
    float* out_f = (float*)d_out;
    k_gate_partial<<<32, 256, 0, stream>>>(X3, gate_w, label, eps, GPART);
    k_gate_final<<<1, 32, 0, stream>>>(GPART, out_f + (size_t)NTOK * 768, GATES);

    // ---- experts 0,1: d_out = X3 + g0*y0 + g1*y1
    cvt(X3, X3BF, (size_t)NTOK * 768);
    for (int k = 0; k < 2; ++k) {
        k_gemm_bf16_wmma<19><<<dim3(384 / 128, NTOK / 128), 256, 0, stream>>>(
            X3BF, WE1 + (size_t)k * 768 * 384, e1b + k * 384,
            nullptr, nullptr, EXPH, NTOK, 384, 768);           // gelu(x@W1+b1) bf16
        k_gemm_bf16_wmma<13><<<dim3(768 / 128, NTOK / 128), 256, 0, stream>>>(
            EXPH, WE2 + (size_t)k * 384 * 768, e2b + k * 768,
            (k == 0) ? X3 : out_f, GATES + k, out_f, NTOK, 768, 384);
    }
}
